// STGCNModel_4922032521320
// MI455X (gfx1250) — matrix operand order repaired and checked by reading, not compiled, not run on hardware
//
#include <hip/hip_runtime.h>
#include <hip/hip_bf16.h>
#include <math.h>

typedef _Float16 half16 __attribute__((ext_vector_type(16)));
typedef float    f32x8  __attribute__((ext_vector_type(8)));
typedef unsigned int u32x4 __attribute__((ext_vector_type(4)));

#define EPS_  1e-5f
#define B_    64
#define T_    256
#define N_    33
#define TN_   8448      /* T_*N_ */
#define MTOT_ 540672    /* B_*T_*N_ (multiple of 128) */

union FragU { half16 h; u32x4 u[2]; };

__device__ inline half16 frag_zero() {
  FragU f;
#pragma unroll
  for (int i = 0; i < 16; ++i) f.h[i] = (_Float16)0.0f;
  return f.h;
}

// A fragment (16x32 f16, MxK): per-lane halves at [p..p+7] and [p+16..p+23]
__device__ inline half16 load_a_frag(const _Float16* p) {
  FragU f;
  f.u[0] = *(const u32x4*)(p);
  f.u[1] = *(const u32x4*)(p + 16);
  return f.h;
}

// B fragment (32x16 f16, KxN): packed fragment-major, 16 contiguous halves per lane
__device__ inline half16 load_b_frag(const _Float16* p) {
  FragU f;
  f.u[0] = *(const u32x4*)(p);
  f.u[1] = *(const u32x4*)(p + 8);
  return f.h;
}

__device__ inline f32x8 wmma16(half16 a, half16 b, f32x8 c) {
  return __builtin_amdgcn_wmma_f32_16x16x32_f16(false, a, false, b, (short)0, c,
                                                false, false);
}

// ---------------------------------------------------------------------------
// Weight pre-pack into B-fragment-major f16 layout.
// dst[((ktg*NT + nt)*32 + lane)*16 + e] = W[k = ktg*32 + (e&8)*2 + (lane/16)*8 + (e&7)]
//                                          [n = nt*16 + lane%16]
// modes: 0 = tcn (src[N][I][9], k = tap*I + i), 1 = gcn (src[K][N]),
//        2 = res 1x1 (src[N][K])
__global__ void pack_b_kernel(const float* __restrict__ src,
                              _Float16* __restrict__ dst,
                              int Ktot, int Nout, int I, int mode) {
  int tid = blockIdx.x * blockDim.x + threadIdx.x;
  int total = Ktot * Nout;
  if (tid >= total) return;
  int e   = tid & 15;
  int l   = (tid >> 4) & 31;
  int t2  = tid >> 9;
  int nt16 = Nout >> 4;
  int nt  = t2 % nt16;
  int ktg = t2 / nt16;
  int k = ktg * 32 + (e & 8) * 2 + (l >> 4) * 8 + (e & 7);
  int n = nt * 16 + (l & 15);
  float v;
  if (mode == 0)      { int tap = k / I; int i = k - tap * I; v = src[(n * I + i) * 9 + tap]; }
  else if (mode == 1) { v = src[k * Nout + n]; }
  else                { v = src[n * I + k]; }
  dst[tid] = (_Float16)v;
}

__global__ void zero_f32(float* __restrict__ p, int n) {
  int i = blockIdx.x * blockDim.x + threadIdx.x;
  if (i < n) p[i] = 0.f;
}

// ---------------------------------------------------------------------------
// K1: input BN + gcn1 (K=3, scalar) + adj aggregation + bias + ReLU -> S1 f16
__global__ void k1_bn_gcn1(const float* __restrict__ x, const float* __restrict__ adj,
                           const float* __restrict__ g_in, const float* __restrict__ b_in,
                           const float* __restrict__ m_in, const float* __restrict__ v_in,
                           const float* __restrict__ gw1, const float* __restrict__ gb1,
                           _Float16* __restrict__ S1) {
  __shared__ float xbn[N_ * 3];
  __shared__ float sup[N_ * 64];
  __shared__ float adjs[N_ * N_];
  int bt = blockIdx.x;
  int tid = threadIdx.x;
  if (tid < N_ * 3) {
    int c = tid % 3;
    float sc = g_in[c] * rsqrtf(v_in[c] + EPS_);
    xbn[tid] = (x[(size_t)bt * (N_ * 3) + tid] - m_in[c]) * sc + b_in[c];
  }
  for (int i = tid; i < N_ * N_; i += blockDim.x) adjs[i] = adj[i];
  __syncthreads();
  for (int i = tid; i < N_ * 64; i += blockDim.x) {
    int n = i >> 6, o = i & 63;
    float s = 0.f;
#pragma unroll
    for (int c = 0; c < 3; ++c) s += xbn[n * 3 + c] * gw1[c * 64 + o];
    sup[i] = s;
  }
  __syncthreads();
  for (int i = tid; i < N_ * 64; i += blockDim.x) {
    int u = i >> 6, o = i & 63;
    float acc = gb1[o];
    for (int v = 0; v < N_; ++v) acc += adjs[u * N_ + v] * sup[v * 64 + o];
    S1[((size_t)bt * N_ + u) * 64 + o] = (_Float16)fmaxf(acc, 0.f);
  }
}

// ---------------------------------------------------------------------------
// K2: tcn1 (9-tap WMMA, K=64 per tap) + bias + BN1 + ReLU + res1 -> out1 f16
__global__ void k2_tcn1(const _Float16* __restrict__ S1, const _Float16* __restrict__ W1p,
                        const float* __restrict__ x,
                        const float* __restrict__ g_in, const float* __restrict__ b_in,
                        const float* __restrict__ m_in, const float* __restrict__ v_in,
                        const float* __restrict__ tb1,
                        const float* __restrict__ g1, const float* __restrict__ b1,
                        const float* __restrict__ m1, const float* __restrict__ v1,
                        const float* __restrict__ rw1, const float* __restrict__ rb1,
                        _Float16* __restrict__ out1) {
  int wave = threadIdx.x >> 5;
  int lane = threadIdx.x & 31;
  int g = lane >> 4, ln = lane & 15;
  long row0 = ((long)blockIdx.x * 8 + wave) * 16;
  long myrow = row0 + ln;
  int myt = (int)((myrow / N_) % T_);

  f32x8 acc[4];
#pragma unroll
  for (int nt = 0; nt < 4; ++nt)
#pragma unroll
    for (int v = 0; v < 8; ++v) acc[nt][v] = 0.f;

  for (int tap = 0; tap < 9; ++tap) {
    int dt = tap - 4;
    bool valid = ((unsigned)(myt + dt)) < (unsigned)T_;
    long ar = valid ? (myrow + (long)dt * N_) : myrow;
    const _Float16* arow = S1 + ar * 64;
#pragma unroll
    for (int kt = 0; kt < 2; ++kt) {
      half16 a = valid ? load_a_frag(arow + kt * 32 + g * 8) : frag_zero();
      int ktg = tap * 2 + kt;
#pragma unroll
      for (int nt = 0; nt < 4; ++nt) {
        half16 b = load_b_frag(W1p + ((size_t)(ktg * 4 + nt) * 32 + lane) * 16);
        acc[nt] = wmma16(a, b, acc[nt]);
      }
    }
  }

  // recompute residual basis (BN'd input, 3 channels) for this lane's 8 rows
  float sc_in[3];
#pragma unroll
  for (int c = 0; c < 3; ++c) sc_in[c] = g_in[c] * rsqrtf(v_in[c] + EPS_);
  float xb[8][3];
#pragma unroll
  for (int v = 0; v < 8; ++v) {
    long r = row0 + g * 8 + v;
#pragma unroll
    for (int c = 0; c < 3; ++c)
      xb[v][c] = (x[r * 3 + c] - m_in[c]) * sc_in[c] + b_in[c];
  }

#pragma unroll
  for (int nt = 0; nt < 4; ++nt) {
    int ch = nt * 16 + ln;
    float sc = g1[ch] * rsqrtf(v1[ch] + EPS_);
    float mm = m1[ch], bb = b1[ch], tb = tb1[ch];
    float w0 = rw1[ch * 3 + 0], w1 = rw1[ch * 3 + 1], w2 = rw1[ch * 3 + 2];
    float rb = rb1[ch];
#pragma unroll
    for (int v = 0; v < 8; ++v) {
      long r = row0 + g * 8 + v;
      float val = acc[nt][v] + tb;
      val = fmaxf((val - mm) * sc + bb, 0.f);
      float res = xb[v][0] * w0 + xb[v][1] * w1 + xb[v][2] * w2 + rb;
      out1[r * 64 + ch] = (_Float16)(val + res);
    }
  }
}

// ---------------------------------------------------------------------------
// K3: gcn2 (WMMA, K=64 -> 128ch) per (b,t), adj aggregation in LDS, ReLU -> S2
__global__ void k3_gcn2(const _Float16* __restrict__ out1, const _Float16* __restrict__ G2p,
                        const float* __restrict__ adj, const float* __restrict__ gb2,
                        _Float16* __restrict__ S2) {
  __shared__ float sup[N_ * 128];
  __shared__ float adjs[N_ * N_];
  int bt = blockIdx.x;
  int tid = threadIdx.x;
  int wave = tid >> 5, lane = tid & 31;
  int g = lane >> 4, ln = lane & 15;
  for (int i = tid; i < N_ * N_; i += blockDim.x) adjs[i] = adj[i];

  long base = (long)bt * N_;
  int nt = wave;  // 8 waves cover the 8 column tiles of 128 channels
  for (int mt = 0; mt < 3; ++mt) {
    int m = mt * 16 + ln;
    bool valid = m < N_;
    const _Float16* arow = out1 + (base + (valid ? m : 0)) * 64;
    f32x8 acc;
#pragma unroll
    for (int v = 0; v < 8; ++v) acc[v] = 0.f;
#pragma unroll
    for (int kt = 0; kt < 2; ++kt) {
      half16 a = valid ? load_a_frag(arow + kt * 32 + g * 8) : frag_zero();
      half16 b = load_b_frag(G2p + ((size_t)(kt * 8 + nt) * 32 + lane) * 16);
      acc = wmma16(a, b, acc);
    }
    int col = nt * 16 + ln;
#pragma unroll
    for (int v = 0; v < 8; ++v) {
      int mm = mt * 16 + g * 8 + v;
      if (mm < N_) sup[mm * 128 + col] = acc[v];
    }
  }
  __syncthreads();
  for (int i = tid; i < N_ * 128; i += blockDim.x) {
    int u = i >> 7, o = i & 127;
    float a2 = gb2[o];
    for (int v = 0; v < N_; ++v) a2 += adjs[u * N_ + v] * sup[v * 128 + o];
    S2[(base + u) * 128 + o] = (_Float16)fmaxf(a2, 0.f);
  }
}

// ---------------------------------------------------------------------------
// K4: tcn2 (9-tap WMMA, K=128/tap) + BN2/ReLU + res2 (WMMA K=64) + fused pool
__global__ void k4_tcn2(const _Float16* __restrict__ S2, const _Float16* __restrict__ W2p,
                        const _Float16* __restrict__ out1, const _Float16* __restrict__ R2p,
                        const float* __restrict__ tb2,
                        const float* __restrict__ g2, const float* __restrict__ b2,
                        const float* __restrict__ m2, const float* __restrict__ v2,
                        const float* __restrict__ rb2,
                        float* __restrict__ poolg) {
  __shared__ float pool[128];
  int tid = threadIdx.x;
  if (tid < 128) pool[tid] = 0.f;
  __syncthreads();
  int wave = tid >> 5, lane = tid & 31;
  int g = lane >> 4, ln = lane & 15;
  long row0 = ((long)blockIdx.x * 8 + wave) * 16;
  long myrow = row0 + ln;
  int myt = (int)((myrow / N_) % T_);
  int b0 = (int)(((long)blockIdx.x * 128) / TN_);  // 8448 = 66*128: single-b block

  f32x8 acc[8];
#pragma unroll
  for (int nt = 0; nt < 8; ++nt)
#pragma unroll
    for (int v = 0; v < 8; ++v) acc[nt][v] = 0.f;

  for (int tap = 0; tap < 9; ++tap) {
    int dt = tap - 4;
    bool valid = ((unsigned)(myt + dt)) < (unsigned)T_;
    long ar = valid ? (myrow + (long)dt * N_) : myrow;
    const _Float16* arow = S2 + ar * 128;
#pragma unroll
    for (int kt = 0; kt < 4; ++kt) {
      half16 a = valid ? load_a_frag(arow + kt * 32 + g * 8) : frag_zero();
      int ktg = tap * 4 + kt;
#pragma unroll
      for (int nt = 0; nt < 8; ++nt) {
        half16 b = load_b_frag(W2p + ((size_t)(ktg * 8 + nt) * 32 + lane) * 16);
        acc[nt] = wmma16(a, b, acc[nt]);
      }
    }
  }

  const _Float16* rrow = out1 + myrow * 64;
#pragma unroll
  for (int nt = 0; nt < 8; ++nt) {
    f32x8 accR;
#pragma unroll
    for (int v = 0; v < 8; ++v) accR[v] = 0.f;
#pragma unroll
    for (int kt = 0; kt < 2; ++kt) {
      half16 a = load_a_frag(rrow + kt * 32 + g * 8);
      half16 b = load_b_frag(R2p + ((size_t)(kt * 8 + nt) * 32 + lane) * 16);
      accR = wmma16(a, b, accR);
    }
    int ch = nt * 16 + ln;
    float sc = g2[ch] * rsqrtf(v2[ch] + EPS_);
    float mm = m2[ch], bb = b2[ch], tb = tb2[ch], rb = rb2[ch];
    float part = 0.f;
#pragma unroll
    for (int v = 0; v < 8; ++v) {
      float val = acc[nt][v] + tb;
      val = fmaxf((val - mm) * sc + bb, 0.f);
      part += val + accR[v] + rb;
    }
    atomicAdd(&pool[ch], part);
  }
  __syncthreads();
  if (tid < 128) atomicAdd(&poolg[b0 * 128 + tid], pool[tid]);
}

// ---------------------------------------------------------------------------
// K5: mean pool finalize + FC(128->10) + sigmoid
__global__ void k5_head(const float* __restrict__ poolg, const float* __restrict__ fcw,
                        const float* __restrict__ fcb, float* __restrict__ out) {
  __shared__ float p[128];
  int b = blockIdx.x, tid = threadIdx.x;
  if (tid < 128) p[tid] = poolg[b * 128 + tid] * (1.f / (float)TN_);
  __syncthreads();
  if (tid < 10) {
    float a = fcb[tid];
    for (int c = 0; c < 128; ++c) a += p[c] * fcw[c * 10 + tid];
    out[b * 10 + tid] = 1.f / (1.f + expf(-a));
  }
}

// ---------------------------------------------------------------------------
extern "C" void kernel_launch(void* const* d_in, const int* in_sizes, int n_in,
                              void* d_out, int out_size, void* d_ws, size_t ws_size,
                              hipStream_t stream) {
  (void)in_sizes; (void)n_in; (void)out_size; (void)ws_size;
  const float* x    = (const float*)d_in[0];
  const float* adj  = (const float*)d_in[1];
  const float* g_in = (const float*)d_in[2];
  const float* b_in = (const float*)d_in[3];
  const float* m_in = (const float*)d_in[4];
  const float* v_in = (const float*)d_in[5];
  const float* gw1  = (const float*)d_in[6];
  const float* gb1  = (const float*)d_in[7];
  const float* tw1  = (const float*)d_in[8];
  const float* tb1  = (const float*)d_in[9];
  const float* g1   = (const float*)d_in[10];
  const float* b1   = (const float*)d_in[11];
  const float* m1   = (const float*)d_in[12];
  const float* v1   = (const float*)d_in[13];
  const float* rw1  = (const float*)d_in[14];
  const float* rb1  = (const float*)d_in[15];
  const float* gw2  = (const float*)d_in[16];
  const float* gb2  = (const float*)d_in[17];
  const float* tw2  = (const float*)d_in[18];
  const float* tb2  = (const float*)d_in[19];
  const float* g2   = (const float*)d_in[20];
  const float* b2   = (const float*)d_in[21];
  const float* m2   = (const float*)d_in[22];
  const float* v2   = (const float*)d_in[23];
  const float* rw2  = (const float*)d_in[24];
  const float* rb2  = (const float*)d_in[25];
  const float* fcw  = (const float*)d_in[26];
  const float* fcb  = (const float*)d_in[27];
  float* out = (float*)d_out;

  char* ws = (char*)d_ws;
  size_t off = 0;
  auto wsalloc = [&](size_t bytes) -> char* {
    char* p = ws + off;
    off += (bytes + 255) & ~(size_t)255;
    return p;
  };
  _Float16* S1  = (_Float16*)wsalloc((size_t)MTOT_ * 64 * 2);
  _Float16* O1  = (_Float16*)wsalloc((size_t)MTOT_ * 64 * 2);
  _Float16* S2  = (_Float16*)wsalloc((size_t)MTOT_ * 128 * 2);
  _Float16* W1p = (_Float16*)wsalloc((size_t)9 * 64 * 64 * 2);
  _Float16* W2p = (_Float16*)wsalloc((size_t)9 * 128 * 128 * 2);
  _Float16* G2p = (_Float16*)wsalloc((size_t)64 * 128 * 2);
  _Float16* R2p = (_Float16*)wsalloc((size_t)64 * 128 * 2);
  float*    poolg = (float*)wsalloc((size_t)B_ * 128 * 4);

  zero_f32<<<(B_ * 128 + 255) / 256, 256, 0, stream>>>(poolg, B_ * 128);
  pack_b_kernel<<<(9 * 64 * 64 + 255) / 256, 256, 0, stream>>>(tw1, W1p, 9 * 64, 64, 64, 0);
  pack_b_kernel<<<(9 * 128 * 128 + 255) / 256, 256, 0, stream>>>(tw2, W2p, 9 * 128, 128, 128, 0);
  pack_b_kernel<<<(64 * 128 + 255) / 256, 256, 0, stream>>>(gw2, G2p, 64, 128, 64, 1);
  pack_b_kernel<<<(64 * 128 + 255) / 256, 256, 0, stream>>>(rw2, R2p, 64, 128, 64, 2);

  k1_bn_gcn1<<<B_ * T_, 256, 0, stream>>>(x, adj, g_in, b_in, m_in, v_in, gw1, gb1, S1);
  k2_tcn1<<<MTOT_ / 128, 256, 0, stream>>>(S1, W1p, x, g_in, b_in, m_in, v_in,
                                           tb1, g1, b1, m1, v1, rw1, rb1, O1);
  k3_gcn2<<<B_ * T_, 256, 0, stream>>>(O1, G2p, adj, gb2, S2);
  k4_tcn2<<<MTOT_ / 128, 256, 0, stream>>>(S2, W2p, O1, R2p, tb2, g2, b2, m2, v2,
                                           rb2, poolg);
  k5_head<<<B_, 128, 0, stream>>>(poolg, fcw, fcb, out);
}
